// NeighborhoodGraphModule_36807869727160
// MI455X (gfx1250) — compile-verified, matrix operand order checked by reading, not run on hardware
//
#include <hip/hip_runtime.h>
#include <hip/hip_bf16.h>

typedef __attribute__((ext_vector_type(16))) _Float16 v16h;
typedef __attribute__((ext_vector_type(8)))  float    v8f;

#define DIMD   256
#define KNBR   32
#define HDIM   512
#define NLAYER 2
#define LSMAXC 0.2f
#define H16S   520   // halves stride for h (multiple of 8 -> 16B aligned rows)
#define E16S   264   // halves stride for encoded
#define BUFS   264   // float stride for f32 scratch

union FragU { v16h v; _Float16 h[16]; float4 q4[2]; };

// 16-bit A/B fragment load (16x32 A layout, per ISA 7.12.2):
// lanes 0-15: row = lane, k halves {0..7} and {16..23}
// lanes 16-31: row = lane-16, k halves {8..15} and {24..31}
// B uses identical pattern from N-major (i.e. [n][k]) memory.
static __device__ __forceinline__ v16h load_frag16(const _Float16* p, int strideH) {
  const int lane = threadIdx.x & 31;
  const int r    = lane & 15;
  const int hh   = (lane >> 4) << 3;          // 0 or 8 halves
  const _Float16* q = p + r * strideH + hh;
  FragU u;
  u.q4[0] = *(const float4*)(q);
  u.q4[1] = *(const float4*)(q + 16);
  return u.v;
}

// C layout: VGPR j, lanes 0-15 -> M=j, N=lane; lanes 16-31 -> M=8+j, N=lane-16
static __device__ __forceinline__ void store_fragC(float* base, int stride, v8f c, float bias) {
  const int lane = threadIdx.x & 31;
  const int col  = lane & 15;
  const int r0   = (lane >> 4) << 3;
#pragma unroll
  for (int j = 0; j < 8; ++j) base[(r0 + j) * stride + col] = c[j] + bias;
}

// out[32 x 256] = A(32 x 32*ksteps, LDS f16) * B(32*ksteps x 256, global f16
// N-major) + bias.  Each wave owns a 32x32 output block (2x2 register tiles):
// 4 independent WMMAs per k-step sharing 2 A-frags (LDS) and 2 B-frags (global).
static __device__ __forceinline__ void wmma_32xD(const _Float16* Alds, int strideA,
                                                 const _Float16* Bg, int strideB,
                                                 int ksteps, float* outb,
                                                 const float* bias) {
  const int wv   = threadIdx.x >> 5;
  const int lane = threadIdx.x & 31;
  const int nt0  = wv * 2;
  const int nt1  = wv * 2 + 1;
  v8f a00 = {}, a01 = {}, a10 = {}, a11 = {};   // acc[mt][ni]
#pragma unroll 2
  for (int kk = 0; kk < ksteps; ++kk) {
    v16h fa0 = load_frag16(Alds + kk * 32, strideA);
    v16h fa1 = load_frag16(Alds + 16 * strideA + kk * 32, strideA);
    v16h fb0 = load_frag16(Bg + (nt0 * 16) * strideB + kk * 32, strideB);
    v16h fb1 = load_frag16(Bg + (nt1 * 16) * strideB + kk * 32, strideB);
    a00 = __builtin_amdgcn_wmma_f32_16x16x32_f16(false, fa0, false, fb0, (short)0, a00, false, false);
    a10 = __builtin_amdgcn_wmma_f32_16x16x32_f16(false, fa1, false, fb0, (short)0, a10, false, false);
    a01 = __builtin_amdgcn_wmma_f32_16x16x32_f16(false, fa0, false, fb1, (short)0, a01, false, false);
    a11 = __builtin_amdgcn_wmma_f32_16x16x32_f16(false, fa1, false, fb1, (short)0, a11, false, false);
  }
  const float bv0 = bias[nt0 * 16 + (lane & 15)];
  const float bv1 = bias[nt1 * 16 + (lane & 15)];
  store_fragC(outb + nt0 * 16,             BUFS, a00, bv0);
  store_fragC(outb + 16 * BUFS + nt0 * 16, BUFS, a10, bv0);
  store_fragC(outb + nt1 * 16,             BUFS, a01, bv1);
  store_fragC(outb + 16 * BUFS + nt1 * 16, BUFS, a11, bv1);
}

// one-time weight conversion: kw/vw -> f16 (layout kept: [l][d][e], N-major for B),
// mlp_w2 [H][D] -> w2T f16 [D][H]
__global__ __launch_bounds__(256)
void ngm_prep(const float* __restrict__ kw, const float* __restrict__ vw,
              const float* __restrict__ w2,
              _Float16* __restrict__ kw16, _Float16* __restrict__ vw16,
              _Float16* __restrict__ w2T16) {
  int i = blockIdx.x * 256 + threadIdx.x;
  if (i < NLAYER * DIMD * DIMD) {
    kw16[i] = (_Float16)kw[i];
    vw16[i] = (_Float16)vw[i];
  }
  if (i < DIMD * HDIM) {
    int d = i >> 9, e = i & (HDIM - 1);
    w2T16[i] = (_Float16)w2[e * DIMD + d];
  }
}

__global__ __launch_bounds__(256)
void ngm_main(const float* __restrict__ tf, const float* __restrict__ nf,
              const int* __restrict__ maskp, const float* __restrict__ relpos,
              const float* __restrict__ dists,
              const float* __restrict__ tn_w, const float* __restrict__ tn_b,
              const float* __restrict__ nn_w, const float* __restrict__ nn_b,
              const float* __restrict__ qw, const float* __restrict__ qb,
              const float* __restrict__ kb_, const float* __restrict__ vb_,
              const float* __restrict__ ow, const float* __restrict__ ob,
              const float* __restrict__ ls, const float* __restrict__ w1,
              const float* __restrict__ b1, const float* __restrict__ b2,
              const float* __restrict__ dist_scale,
              const float* __restrict__ gate_w, const float* __restrict__ gate_b,
              const _Float16* __restrict__ kw16, const _Float16* __restrict__ vw16,
              const _Float16* __restrict__ w2T16,
              float* __restrict__ ogf, float* __restrict__ ogate,
              float* __restrict__ oattn, float* __restrict__ oent,
              float* __restrict__ ovc) {
  __shared__ __align__(16) _Float16 h16[KNBR * H16S];   // 33280 B
  __shared__ __align__(16) _Float16 e16[KNBR * E16S];   // 16896 B
  __shared__ __align__(16) float bufA[KNBR * BUFS];     // 33792 B
  __shared__ __align__(16) float tgt[DIMD], gf[DIMD], lnx[DIMD], qv[DIMD], agg[DIMD];
  __shared__ float rp0[KNBR], rp1[KNBR], dst[KNBR], attn[KNBR], lgt[KNBR];
  __shared__ int   mskv[KNBR];
  __shared__ float red1[8], red2[8];
  __shared__ float s_vc;
  __shared__ int   s_nov;

  const int n = blockIdx.x;
  const int t = threadIdx.x;

  // ---- phase 0: per-node loads -------------------------------------------
  {
    float tv = tf[(size_t)n * DIMD + t];
    tgt[t] = tv;
    gf[t]  = tv;
  }
  if (t < KNBR) {
    int mk = (maskp[(size_t)n * KNBR + t] != 0);
    mskv[t] = mk;
    unsigned long long bal = __ballot(mk);   // wave 0 only (t<32)
    if (t == 0) { s_vc = (float)__popcll(bal); s_nov = (bal == 0ull); }
    rp0[t] = relpos[((size_t)n * KNBR + t) * 2 + 0];
    rp1[t] = relpos[((size_t)n * KNBR + t) * 2 + 1];
    dst[t] = dists[(size_t)n * KNBR + t];
  }
  __syncthreads();

  const float dscale = fabsf(dist_scale[0]);

  // ---- phase 1: h = GELU(df @ w1 + b1), f16 into LDS ---------------------
#pragma unroll 4
  for (int rep = 0; rep < (KNBR * HDIM) / 256; ++rep) {
    int idx = rep * 256 + t;
    int k = idx >> 9, e = idx & (HDIM - 1);
    float x = rp0[k] * w1[e] + rp1[k] * w1[HDIM + e] + dst[k] * w1[2 * HDIM + e] + b1[e];
    float g = 0.5f * x * (1.0f + erff(x * 0.70710678118654752f));
    h16[k * H16S + e] = (_Float16)g;
  }
  __syncthreads();

  // ---- phase 2: pos_enc = h @ w2 + b2 (WMMA) -----------------------------
  wmma_32xD(h16, H16S, w2T16, HDIM, HDIM / 32, bufA, b2);
  __syncthreads();

  // ---- phase 3: encoded = LN(nbr + pos_enc), f16 into LDS ----------------
  {
    int row = t >> 3, sub = t & 7;
    const float* nrow = nf + ((size_t)n * KNBR + row) * DIMD;
    float s1 = 0.f, s2 = 0.f;
#pragma unroll
    for (int j = 0; j < DIMD / 8; ++j) {
      int d = sub + 8 * j;
      float x = nrow[d] + bufA[row * BUFS + d];
      bufA[row * BUFS + d] = x;
      s1 += x; s2 += x * x;
    }
    for (int m = 4; m >= 1; m >>= 1) { s1 += __shfl_xor(s1, m, 8); s2 += __shfl_xor(s2, m, 8); }
    float mean = s1 * (1.0f / DIMD);
    float var  = s2 * (1.0f / DIMD) - mean * mean;
    float inv  = rsqrtf(var + 1e-5f);
#pragma unroll
    for (int j = 0; j < DIMD / 8; ++j) {
      int d = sub + 8 * j;
      float x = (bufA[row * BUFS + d] - mean) * inv * nn_w[d] + nn_b[d];
      e16[row * E16S + d] = (_Float16)x;
    }
  }
  __syncthreads();

  // ---- layer loop ---------------------------------------------------------
  for (int l = 0; l < NLAYER; ++l) {
    // a) LN(gf) across block
    float x = gf[t];
    float s1 = x, s2 = x * x;
    for (int m = 16; m >= 1; m >>= 1) { s1 += __shfl_xor(s1, m, 32); s2 += __shfl_xor(s2, m, 32); }
    if ((t & 31) == 0) { red1[t >> 5] = s1; red2[t >> 5] = s2; }
    __syncthreads();
    float S1 = 0.f, S2 = 0.f;
#pragma unroll
    for (int i = 0; i < 8; ++i) { S1 += red1[i]; S2 += red2[i]; }
    float mean = S1 * (1.0f / DIMD), var = S2 * (1.0f / DIMD) - mean * mean;
    float inv = rsqrtf(var + 1e-5f);
    lnx[t] = (x - mean) * inv * tn_w[t] + tn_b[t];
    __syncthreads();

    // b) q = lnx @ qw^T + qb (matvec, f32, per-thread rows)
    {
      const float4* qr = (const float4*)(qw + ((size_t)l * DIMD + t) * DIMD);
      const float4* lx = (const float4*)lnx;
      float a = 0.f;
#pragma unroll 8
      for (int e4 = 0; e4 < DIMD / 4; ++e4) {
        float4 wv4 = qr[e4], xv = lx[e4];
        a += wv4.x * xv.x + wv4.y * xv.y + wv4.z * xv.z + wv4.w * xv.w;
      }
      qv[t] = qb[l * DIMD + t] + a;
    }

    // c) k_ = encoded @ kw^T + kb (WMMA)
    wmma_32xD(e16, E16S, kw16 + (size_t)l * DIMD * DIMD, DIMD, DIMD / 32, bufA,
              kb_ + l * DIMD);
    __syncthreads();

    // d) logits
    {
      int row = t >> 3, sub = t & 7;
      float p = 0.f;
#pragma unroll
      for (int j = 0; j < DIMD / 8; ++j) { int d = sub + 8 * j; p += qv[d] * bufA[row * BUFS + d]; }
      for (int m = 4; m >= 1; m >>= 1) p += __shfl_xor(p, m, 8);
      if (sub == 0) {
        float lgv = p * 0.0625f - dst[row] * dscale;   // 1/sqrt(256)
        if (!mskv[row]) lgv = -3.402823466e38f;
        if (s_nov)      lgv = 0.0f;
        lgt[row] = lgv;
      }
    }
    __syncthreads();

    // e) softmax (wave 0)
    if (t < 32) {
      float v = lgt[t];
      float mx = v;
      for (int m = 16; m >= 1; m >>= 1) mx = fmaxf(mx, __shfl_xor(mx, m, 32));
      float ev = __expf(v - mx);
      float s = ev;
      for (int m = 16; m >= 1; m >>= 1) s += __shfl_xor(s, m, 32);
      attn[t] = ev / s;
    }
    __syncthreads();

    // f) v_ = encoded @ vw^T + vb (WMMA, reuse bufA)
    wmma_32xD(e16, E16S, vw16 + (size_t)l * DIMD * DIMD, DIMD, DIMD / 32, bufA,
              vb_ + l * DIMD);
    __syncthreads();

    // g) agg[d] = sum_k attn[k] * v_[k,d]
    {
      float a = 0.f;
#pragma unroll
      for (int k = 0; k < KNBR; ++k) a += attn[k] * bufA[k * BUFS + t];
      agg[t] = a;
    }
    __syncthreads();

    // h) out proj + layer-scale residual
    {
      const float4* orow = (const float4*)(ow + ((size_t)l * DIMD + t) * DIMD);
      const float4* ax = (const float4*)agg;
      float a = 0.f;
#pragma unroll 8
      for (int e4 = 0; e4 < DIMD / 4; ++e4) {
        float4 wv4 = orow[e4], xv = ax[e4];
        a += wv4.x * xv.x + wv4.y * xv.y + wv4.z * xv.z + wv4.w * xv.w;
      }
      float o  = ob[l * DIMD + t] + a;
      float gm = fminf(fmaxf(ls[l * DIMD + t], -LSMAXC), LSMAXC);
      gf[t] = gf[t] + o * gm;
    }
    __syncthreads();
  }

  // ---- gate, entropy, outputs --------------------------------------------
  {
    float gp = tgt[t] * gate_w[t];
    for (int m = 16; m >= 1; m >>= 1) gp += __shfl_xor(gp, m, 32);
    if ((t & 31) == 0) red1[t >> 5] = gp;
  }
  __syncthreads();
  float G = 0.f;
#pragma unroll
  for (int i = 0; i < 8; ++i) G += red1[i];
  float gate = 1.0f / (1.0f + __expf(-(G + gate_b[0])));

  if (t < 32) {
    float pa = fmaxf(attn[t], 1e-8f);
    float c = -pa * __logf(pa);
    for (int m = 16; m >= 1; m >>= 1) c += __shfl_xor(c, m, 32);
    oattn[(size_t)n * KNBR + t] = attn[t];
    if (t == 0) { oent[n] = c; ogate[n] = gate; ovc[n] = s_vc; }
  }
  ogf[(size_t)n * DIMD + t] = gf[t] * gate;
}

extern "C" void kernel_launch(void* const* d_in, const int* in_sizes, int n_in,
                              void* d_out, int out_size, void* d_ws, size_t ws_size,
                              hipStream_t stream) {
  const float* tf   = (const float*)d_in[0];
  const float* nf   = (const float*)d_in[1];
  const int*   msk  = (const int*)d_in[2];
  const float* rp   = (const float*)d_in[3];
  const float* dist = (const float*)d_in[4];
  const float* tn_w = (const float*)d_in[5];
  const float* tn_b = (const float*)d_in[6];
  const float* nn_w = (const float*)d_in[7];
  const float* nn_b = (const float*)d_in[8];
  const float* qw   = (const float*)d_in[9];
  const float* qb   = (const float*)d_in[10];
  const float* kw   = (const float*)d_in[11];
  const float* kb   = (const float*)d_in[12];
  const float* vw   = (const float*)d_in[13];
  const float* vb   = (const float*)d_in[14];
  const float* ow   = (const float*)d_in[15];
  const float* ob   = (const float*)d_in[16];
  const float* ls   = (const float*)d_in[17];
  const float* w1   = (const float*)d_in[18];
  const float* b1   = (const float*)d_in[19];
  const float* w2   = (const float*)d_in[20];
  const float* b2   = (const float*)d_in[21];
  const float* dsc  = (const float*)d_in[22];
  const float* gw   = (const float*)d_in[23];
  const float* gb   = (const float*)d_in[24];

  const int N = in_sizes[0] / DIMD;

  _Float16* kw16  = (_Float16*)d_ws;
  _Float16* vw16  = kw16 + (size_t)NLAYER * DIMD * DIMD;
  _Float16* w2T16 = vw16 + (size_t)NLAYER * DIMD * DIMD;

  float* ogf   = (float*)d_out;
  float* ogate = ogf + (size_t)N * DIMD;
  float* oattn = ogate + N;
  float* oent  = oattn + (size_t)N * KNBR;
  float* ovc   = oent + N;

  ngm_prep<<<(NLAYER * DIMD * DIMD + 255) / 256, 256, 0, stream>>>(
      kw, vw, w2, kw16, vw16, w2T16);
  ngm_main<<<N, 256, 0, stream>>>(
      tf, nf, msk, rp, dist, tn_w, tn_b, nn_w, nn_b, qw, qb, kb, vb, ow, ob, ls,
      w1, b1, b2, dsc, gw, gb,
      kw16, vw16, w2T16, ogf, ogate, oattn, oent, ovc);
}
